// BiDirectionalGRU_1975684956396
// MI455X (gfx1250) — matrix operand order repaired and checked by reading, not compiled, not run on hardware
//
#include <hip/hip_runtime.h>
#include <hip/hip_bf16.h>
#include <hip/hip_fp16.h>

// ---------------------------------------------------------------------------
// Bidirectional GRU, B=128 T=512 I=256 H=512, for MI455X (gfx1250, wave32).
// bf16 WMMA (v_wmma_f32_16x16x32_bf16) with fp32 accumulation; async-to-LDS
// (GLOBAL_LOAD_ASYNC_TO_LDS_B128 / ASYNCcnt) for all LDS staging.
// ---------------------------------------------------------------------------

typedef __attribute__((ext_vector_type(16))) __bf16 v16bf;
typedef __attribute__((ext_vector_type(8)))  __bf16 v8bf;
typedef __attribute__((ext_vector_type(8)))  float  v8f;
typedef __attribute__((ext_vector_type(4)))  int    v4i;

#define T_  512
#define B_  128
#define I_  256
#define H_  512
#define G3  1536   // 3*H

// ---- CDNA5 async-to-LDS helpers ------------------------------------------
// Builtin prototype (from hipcc diagnostic): param0 = AS(1) v4i* (global src),
// param1 = AS(3) v4i* (LDS dst), then two integer immediates (offset, cpol).

typedef __attribute__((address_space(1))) v4i* g_v4i_ptr;   // global int4*
typedef __attribute__((address_space(3))) v4i* l_v4i_ptr;   // LDS int4*

__device__ __forceinline__ void async_load_b128(const void* gsrc, void* ldst) {
#if __has_builtin(__builtin_amdgcn_global_load_async_to_lds_b128)
  __builtin_amdgcn_global_load_async_to_lds_b128(
      (g_v4i_ptr)(uintptr_t)gsrc,
      (l_v4i_ptr)(uintptr_t)(unsigned)(uintptr_t)ldst,
      0, 0);
#else
  asm volatile("global_load_async_to_lds_b128 %0, %1, off"
               :: "v"((unsigned)(uintptr_t)ldst),
                  "v"((unsigned long long)(uintptr_t)gsrc)
               : "memory");
#endif
}

#if __has_builtin(__builtin_amdgcn_s_wait_asynccnt)
#define WAIT_ASYNC(n) __builtin_amdgcn_s_wait_asynccnt(n)
#else
#define WAIT_ASYNC(n) asm volatile("s_wait_asynccnt %0" :: "n"(n) : "memory")
#endif

// ---- WMMA helpers ---------------------------------------------------------

__device__ __forceinline__ v8f wmma_bf16(v16bf a, v16bf b, v8f c) {
  // (neg_a, A, neg_b, B, c_mod, C, reuse_a, reuse_b)
  return __builtin_amdgcn_wmma_f32_16x16x32_bf16(false, a, false, b,
                                                 (short)0, c, false, false);
}

// A-matrix 16x32 bf16 fragment from a row-major [rows][ld] bf16 buffer.
// Layout: lane l (l<16): row=l, K = 0..15; lane>=16: row=l-16,
// K = 16..31  => two adjacent 16B chunks per lane (matches frag_b).
__device__ __forceinline__ v16bf frag_a(const __bf16* base, int ld,
                                        int row0, int kb, int lane) {
  const int r = lane & 15, g = lane >> 4;
  const __bf16* p = base + (size_t)(row0 + r) * ld + kb + g * 16;
  v8bf lo = *(const v8bf*)p;
  v8bf hi = *(const v8bf*)(p + 8);
  return __builtin_shufflevector(lo, hi, 0,1,2,3,4,5,6,7,
                                         8,9,10,11,12,13,14,15);
}

// B-matrix 32x16 bf16 fragment from an N-major (transposed) [N][ld] buffer.
// lane l: column n = l&15, K = (l>>4)*16 + 0..15  => one contiguous 32B run.
__device__ __forceinline__ v16bf frag_b(const __bf16* base, int ld,
                                        int nb, int kb, int lane) {
  const int n = lane & 15, g = lane >> 4;
  return *(const v16bf*)(base + (size_t)(nb + n) * ld + kb + g * 16);
}

__device__ __forceinline__ float sigmoidf_(float x) {
  return 1.0f / (1.0f + __expf(-x));
}

// ---- Kernel 1: column-norm constrain Wh, transpose to N-major bf16 --------
__global__ __launch_bounds__(256) void prep_wh_kernel(
    const float* __restrict__ Wh_f, const float* __restrict__ Wh_b,
    __bf16* __restrict__ WhT_f, __bf16* __restrict__ WhT_b) {
  const int dir = blockIdx.y;
  const float* Wh = dir ? Wh_b : Wh_f;
  __bf16* WhT = dir ? WhT_b : WhT_f;
  const int j = blockIdx.x;            // column 0..1535
  const int tid = threadIdx.x;
  __shared__ float red[256];
  __shared__ float scale_s;
  float s = 0.0f;
  for (int i = tid; i < H_; i += 256) {
    float w = Wh[(size_t)i * G3 + j];
    s += w * w;
  }
  red[tid] = s;
  __syncthreads();
  for (int ofs = 128; ofs > 0; ofs >>= 1) {
    if (tid < ofs) red[tid] += red[tid + ofs];
    __syncthreads();
  }
  if (tid == 0) {
    float norm = sqrtf(red[0]);
    float desired = fminf(norm, 1.0f);
    scale_s = desired / (1e-7f + norm);
  }
  __syncthreads();
  const float sc = scale_s;
  for (int i = tid; i < H_; i += 256)
    WhT[(size_t)j * H_ + i] = (__bf16)(Wh[(size_t)i * G3 + j] * sc);
}

// ---- Kernel 2: transpose Wx (I,3H) -> WxT (3H,I) bf16 ---------------------
__global__ __launch_bounds__(256) void prep_wx_kernel(
    const float* __restrict__ Wx_f, const float* __restrict__ Wx_b,
    __bf16* __restrict__ WxT_f, __bf16* __restrict__ WxT_b) {
  const int dir = blockIdx.y;
  const float* Wx = dir ? Wx_b : Wx_f;
  __bf16* WxT = dir ? WxT_b : WxT_f;
  const int n = blockIdx.x;            // 0..1535
  const int k = threadIdx.x;           // 0..255
  WxT[(size_t)n * I_ + k] = (__bf16)(Wx[(size_t)k * G3 + n]);
}

// ---- Kernel 3: x fp32 -> bf16 ---------------------------------------------
__global__ __launch_bounds__(256) void convert_x_kernel(
    const float* __restrict__ x, __bf16* __restrict__ xbf) {
  const size_t i = ((size_t)blockIdx.x * 256 + threadIdx.x) * 4;
  float4 v = *(const float4*)(x + i);
  xbf[i + 0] = (__bf16)v.x;
  xbf[i + 1] = (__bf16)v.y;
  xbf[i + 2] = (__bf16)v.z;
  xbf[i + 3] = (__bf16)v.w;
}

// ---- Kernel 4: xp GEMM  xp[t][b][g] = (x @ Wx)[b*T+t][g] + bias[g] --------
// 128x128 tile per block, K=256.  Double-buffered LDS tiles filled with
// async-to-LDS b128 loads so the next K-slice streams in under the WMMAs.
__global__ __launch_bounds__(256) void xp_gemm_kernel(
    const __bf16* __restrict__ xbf,
    const __bf16* __restrict__ WxT_f, const __bf16* __restrict__ WxT_b,
    const float* __restrict__ bias_f, const float* __restrict__ bias_b,
    _Float16* __restrict__ xp_f, _Float16* __restrict__ xp_b) {
  const int dir = blockIdx.z;
  const __bf16* WxT = dir ? WxT_b : WxT_f;
  const float* bias = dir ? bias_b : bias_f;
  _Float16* xp = dir ? xp_b : xp_f;

  const int mb = blockIdx.x * 128;
  const int nb = blockIdx.y * 128;
  const int tid = threadIdx.x, lane = tid & 31, wave = tid >> 5;
  const int wm = wave >> 1;            // 0..3 (M)
  const int wn = wave & 1;             // 0..1 (N)

  __shared__ __bf16 As[2][128 * 32];   // 2 x 8 KB, K-major rows
  __shared__ __bf16 Bs[2][128 * 32];   // 2 x 8 KB, N-major rows

  v8f acc[2][4];
#pragma unroll
  for (int mt = 0; mt < 2; ++mt)
#pragma unroll
    for (int nt = 0; nt < 4; ++nt) acc[mt][nt] = (v8f){};

  const int ldrow = tid >> 1;          // 0..127
  const int ldoff = (tid & 1) * 16;    // 0 or 16 elems
  const __bf16* gA = xbf + (size_t)(mb + ldrow) * I_ + ldoff;
  const __bf16* gB = WxT + (size_t)(nb + ldrow) * I_ + ldoff;

  // stage K-slice kb into buffer buf: 4 async b128 per thread
#define STAGE(buf, kb)                                                    \
  do {                                                                    \
    async_load_b128(gA + (kb),     &As[buf][ldrow * 32 + ldoff]);         \
    async_load_b128(gA + (kb) + 8, &As[buf][ldrow * 32 + ldoff + 8]);     \
    async_load_b128(gB + (kb),     &Bs[buf][ldrow * 32 + ldoff]);         \
    async_load_b128(gB + (kb) + 8, &Bs[buf][ldrow * 32 + ldoff + 8]);     \
  } while (0)

  STAGE(0, 0);
  for (int it = 0; it < 8; ++it) {
    const int cur = it & 1;
    if (it < 7) {
      STAGE(cur ^ 1, (it + 1) * 32);   // prefetch next slice (in flight)
      WAIT_ASYNC(4);                   // current slice complete
    } else {
      WAIT_ASYNC(0);
    }
    __syncthreads();                   // all waves' staging visible

    v16bf bfr[4];
#pragma unroll
    for (int nt = 0; nt < 4; ++nt)
      bfr[nt] = frag_b(&Bs[cur][0], 32, wn * 64 + nt * 16, 0, lane);
#pragma unroll
    for (int mt = 0; mt < 2; ++mt) {
      v16bf a = frag_a(&As[cur][0], 32, wm * 32 + mt * 16, 0, lane);
#pragma unroll
      for (int nt = 0; nt < 4; ++nt)
        acc[mt][nt] = wmma_bf16(a, bfr[nt], acc[mt][nt]);
    }
    __syncthreads();                   // safe to overwrite buffer `cur`
  }
#undef STAGE

  // epilogue: add bias, convert f16, scatter to (T,B,3H) layout
#pragma unroll
  for (int mt = 0; mt < 2; ++mt) {
#pragma unroll
    for (int nt = 0; nt < 4; ++nt) {
      const int n_g = nb + wn * 64 + nt * 16 + (lane & 15);
      const float bv = bias[n_g];
#pragma unroll
      for (int i = 0; i < 8; ++i) {
        const int m_g = mb + wm * 32 + mt * 16 + i + 8 * (lane >> 4);
        const int b_idx = m_g >> 9;    // /T
        const int t_idx = m_g & 511;   // %T
        xp[((size_t)t_idx * B_ + b_idx) * G3 + n_g] =
            (_Float16)(acc[mt][nt][i] + bv);
      }
    }
  }
}

// ---- Kernel 5: recurrent GRU scan -----------------------------------------
// grid (8, 2): 8 batch tiles of 16 rows x 2 directions; 8 waves/block, each
// wave owns 64 h-columns.  Per step: the contiguous 48 KB xp tile is staged
// to LDS with async b128 loads whose latency hides behind the z/r GEMM.
// The recurrence is batch-independent -> intra-workgroup barriers only.
__global__ __launch_bounds__(256) void gru_scan_kernel(
    const _Float16* __restrict__ xp_f, const _Float16* __restrict__ xp_b,
    const __bf16* __restrict__ WhT_f, const __bf16* __restrict__ WhT_b,
    float* __restrict__ out) {
  const int dir = blockIdx.y;
  const _Float16* xp = dir ? xp_b : xp_f;
  const __bf16* WhT = dir ? WhT_b : WhT_f;
  float* states = out + (size_t)dir * B_ * T_ * H_;
  float* combined = out + (size_t)2 * B_ * T_ * H_;

  const int btile = blockIdx.x * 16;
  const int tid = threadIdx.x, lane = tid & 31, wave = tid >> 5;
  const int cb = wave * 64;            // wave's column base in H
  const int ln = lane & 15, lg = lane >> 4;

  extern __shared__ char smem[];
  _Float16* xs  = (_Float16*)smem;                        // 48 KB xp tile
  __bf16*   hbuf  = (__bf16*)(smem + 16 * G3 * 2);        // 16 KB h_{t-1}
  __bf16*   rhbuf = (__bf16*)(smem + 16 * G3 * 2 + 16 * H_ * 2); // 16 KB r*h

  for (int i = tid; i < 16 * H_; i += 256) hbuf[i] = (__bf16)0.0f;
  v8f h[4];
#pragma unroll
  for (int nt = 0; nt < 4; ++nt) h[nt] = (v8f){};
  __syncthreads();

  for (int t = 0; t < T_; ++t) {
    const int tt = dir ? (T_ - 1 - t) : t;
    const _Float16* xrow = xp + ((size_t)tt * B_ + btile) * G3;
    const bool last = (t == T_ - 1);

    // ---- stage this step's xp tile (16x1536 f16, contiguous 48 KB):
    //      12 coalesced async b128 loads per thread, consumed after phase 1.
#pragma unroll
    for (int j = 0; j < 12; ++j) {
      const int e = (j * 256 + tid) * 8;     // f16 elements (16B chunks)
      async_load_b128(xrow + e, xs + e);
    }

    // ---- phase 1: z and r gate GEMMs (A from hbuf, B streamed from L2) ----
    v8f zac[4], rac[4];
#pragma unroll
    for (int nt = 0; nt < 4; ++nt) { zac[nt] = (v8f){}; rac[nt] = (v8f){}; }
    for (int kb = 0; kb < H_; kb += 32) {
      v16bf a = frag_a(hbuf, H_, 0, kb, lane);
#pragma unroll
      for (int nt = 0; nt < 4; ++nt) {
        const int nbase = cb + nt * 16;
        v16bf bz = frag_b(WhT, H_, 0 * H_ + nbase, kb, lane);
        v16bf br = frag_b(WhT, H_, 1 * H_ + nbase, kb, lane);
        zac[nt] = wmma_bf16(a, bz, zac[nt]);
        rac[nt] = wmma_bf16(a, br, rac[nt]);
      }
    }
    WAIT_ASYNC(0);
    __syncthreads();   // (B1) xs ready everywhere; all waves done with hbuf

    v8f z[4];
#pragma unroll
    for (int nt = 0; nt < 4; ++nt) {
      const int ncol = cb + nt * 16 + ln;
#pragma unroll
      for (int i = 0; i < 8; ++i) {
        const int m = i + 8 * lg;
        const float xz = (float)xs[m * G3 + 0 * H_ + ncol];
        const float xr = (float)xs[m * G3 + 1 * H_ + ncol];
        const float zz = sigmoidf_(xz + zac[nt][i]);
        const float rr = sigmoidf_(xr + rac[nt][i]);
        z[nt][i] = zz;
        rhbuf[m * H_ + ncol] = (__bf16)(rr * h[nt][i]);
      }
    }
    __syncthreads();   // (B2) rhbuf complete

    // ---- phase 2: candidate GEMM (A from rhbuf), update h -----------------
    v8f hac[4];
#pragma unroll
    for (int nt = 0; nt < 4; ++nt) hac[nt] = (v8f){};
    for (int kb = 0; kb < H_; kb += 32) {
      v16bf a = frag_a(rhbuf, H_, 0, kb, lane);
#pragma unroll
      for (int nt = 0; nt < 4; ++nt) {
        v16bf bh = frag_b(WhT, H_, 2 * H_ + cb + nt * 16, kb, lane);
        hac[nt] = wmma_bf16(a, bh, hac[nt]);
      }
    }
#pragma unroll
    for (int nt = 0; nt < 4; ++nt) {
      const int ncol = cb + nt * 16 + ln;
#pragma unroll
      for (int i = 0; i < 8; ++i) {
        const int m = i + 8 * lg;
        const float xh = (float)xs[m * G3 + 2 * H_ + ncol];
        const float ht = tanhf(xh + hac[nt][i]);
        const float zz = z[nt][i];
        const float hn = (1.0f - zz) * h[nt][i] + zz * ht;
        h[nt][i] = hn;
        hbuf[m * H_ + ncol] = (__bf16)hn;
        states[((size_t)(btile + m) * T_ + tt) * H_ + ncol] = hn;
        if (last)
          combined[(size_t)(btile + m) * (2 * H_) + dir * H_ + ncol] = hn;
      }
    }
    __syncthreads();   // (B3) hbuf ready; xs safe to overwrite next step
  }
}

// ---------------------------------------------------------------------------

extern "C" void kernel_launch(void* const* d_in, const int* in_sizes, int n_in,
                              void* d_out, int out_size, void* d_ws, size_t ws_size,
                              hipStream_t stream) {
  (void)in_sizes; (void)n_in; (void)out_size; (void)ws_size;
  const float* x    = (const float*)d_in[0];   // (B,T,I)
  const float* Wx_f = (const float*)d_in[1];   // (I,3H)
  const float* Wh_f = (const float*)d_in[2];   // (H,3H)
  const float* b_f  = (const float*)d_in[3];   // (3H,)
  const float* Wx_b = (const float*)d_in[4];
  const float* Wh_b = (const float*)d_in[5];
  const float* b_b  = (const float*)d_in[6];
  float* out = (float*)d_out;

  // workspace layout
  char* ws = (char*)d_ws;
  __bf16* xbf   = (__bf16*)ws;  ws += (size_t)B_ * T_ * I_ * 2;   // 32 MB
  __bf16* WxT_f = (__bf16*)ws;  ws += (size_t)G3 * I_ * 2;        // 768 KB
  __bf16* WxT_b = (__bf16*)ws;  ws += (size_t)G3 * I_ * 2;
  __bf16* WhT_f = (__bf16*)ws;  ws += (size_t)G3 * H_ * 2;        // 1.5 MB
  __bf16* WhT_b = (__bf16*)ws;  ws += (size_t)G3 * H_ * 2;
  _Float16* xp_f = (_Float16*)ws; ws += (size_t)T_ * B_ * G3 * 2; // 192 MB
  _Float16* xp_b = (_Float16*)ws; ws += (size_t)T_ * B_ * G3 * 2; // 192 MB

  prep_wh_kernel<<<dim3(G3, 2), 256, 0, stream>>>(Wh_f, Wh_b, WhT_f, WhT_b);
  prep_wx_kernel<<<dim3(G3, 2), 256, 0, stream>>>(Wx_f, Wx_b, WxT_f, WxT_b);
  convert_x_kernel<<<(B_ * T_ * I_) / (256 * 4), 256, 0, stream>>>(x, xbf);
  xp_gemm_kernel<<<dim3((B_ * T_) / 128, G3 / 128, 2), 256, 0, stream>>>(
      xbf, WxT_f, WxT_b, b_f, b_b, xp_f, xp_b);
  // dynamic LDS: 48 KB xp tile + 16 KB h + 16 KB r*h = 80 KB
  gru_scan_kernel<<<dim3(B_ / 16, 2), 256, 80 * 1024, stream>>>(
      xp_f, xp_b, WhT_f, WhT_b, out);
}
